// QKV_MultiHead_Attention_63823214019150
// MI455X (gfx1250) — matrix operand order repaired and checked by reading, not compiled, not run on hardware
//
#include <hip/hip_runtime.h>
#include <hip/hip_bf16.h>

// ---------------------------------------------------------------------------
// MHA forward for MI455X (gfx1250, wave32).  All GEMMs via
// v_wmma_f32_16x16x32_f16.  2x2 register-blocked projection GEMMs,
// 16x2048 f32 score strip in LDS (CDNA5 320KB WGP LDS), softmax probs
// re-packed in place as f16 so the PV A-fragments are 16B LDS loads.
// ---------------------------------------------------------------------------

typedef __attribute__((ext_vector_type(16))) _Float16 v16h;
typedef __attribute__((ext_vector_type(8)))  _Float16 v8h;
typedef __attribute__((ext_vector_type(8)))  float    v8f;

#define B_DIM 2
#define L_DIM 2048
#define D_EMB 1024
#define D_MOD 1024
#define H_DIM 16
#define DK    64
#define NEG_BIG (-9.2233720368547758e18f)
#define LDP   (L_DIM + 4)   // padded LDS row stride (floats)

// ---- fragment loaders (layouts per CDNA5 ISA 7.12.2) ----------------------

// A-matrix 16x32 f16 frag from row-major f16 [rows][ld]
__device__ inline v16h load_a_f16(const _Float16* base, int ld, int row0, int k0, int lane) {
    int M = lane & 15, half = lane >> 4;
    const _Float16* p = base + (size_t)(row0 + M) * ld + k0 + half * 8;
    v8h lo = *(const v8h*)p;         // K = k0 + half*8 + 0..7
    v8h hi = *(const v8h*)(p + 16);  // K = k0 + 16 + half*8 + 0..7
    v16h r;
#pragma unroll
    for (int i = 0; i < 8; ++i) { r[i] = lo[i]; r[i + 8] = hi[i]; }
    return r;
}

// B-matrix 32x16 f16 frag: element i<8 from W[n0+n][k0+khalf*8+i], i>=8 from W[n0+n][k0+16+khalf*8+(i-8)]
__device__ inline v16h load_b_f16(const _Float16* base, int ld, int n0, int k0, int lane) {
    int n = lane & 15, kh = lane >> 4;
    const _Float16* p = base + (size_t)(n0 + n) * ld + k0 + kh * 8;
    v8h lo = *(const v8h*)p;
    v8h hi = *(const v8h*)(p + 16);
    v16h r;
#pragma unroll
    for (int i = 0; i < 8; ++i) { r[i] = lo[i]; r[i + 8] = hi[i]; }
    return r;
}

__device__ inline v8f wmma_f16(v16h a, v16h b, v8f c) {
    return __builtin_amdgcn_wmma_f32_16x16x32_f16(false, a, false, b, (short)0, c, false, false);
}

// ---- f32 -> f16 conversion ------------------------------------------------
__global__ void cvt_f32_f16_kernel(const float* __restrict__ s, _Float16* __restrict__ d, int n) {
    int i = blockIdx.x * blockDim.x + threadIdx.x;
    if (i < n) d[i] = (_Float16)s[i];
}

// ---- tile store for projection GEMM ---------------------------------------
// mode 0: f32 row-major out [M][N]
// mode 1: f16 out  [B,H,L,DK]   (Q / K head-split)
// mode 2: f16 out  [B,H,DK,L]   (V transposed)
__device__ inline void store_tile(void* out, v8f c, int m0, int n0, int N, int mode, int lane) {
    int n = lane & 15, halfm = lane >> 4;
    if (mode == 0) {
        float* o = (float*)out;
#pragma unroll
        for (int j = 0; j < 8; ++j) {
            int M = j + 8 * halfm;
            o[(size_t)(m0 + M) * N + n0 + n] = c[j];
        }
    } else {
        _Float16* o = (_Float16*)out;
        int mcol = n0 + n;
        int h = mcol >> 6, d = mcol & 63;
#pragma unroll
        for (int j = 0; j < 8; ++j) {
            int r  = m0 + j + 8 * halfm;
            int bb = r / L_DIM, l = r % L_DIM;
            size_t idx = (mode == 1)
                ? (((size_t)(bb * H_DIM + h) * L_DIM + l) * DK + d)
                : (((size_t)(bb * H_DIM + h) * DK + d) * L_DIM + l);
            o[idx] = (_Float16)c[j];
        }
    }
}

// ---- generic projection GEMM:  C[M,N] = A[M,K] * W[N,K]^T + bias[N] -------
// 2x2 register blocking: each wave produces a 32x32 tile.
__global__ __launch_bounds__(128)
void gemm_proj_kernel(const _Float16* __restrict__ A, const _Float16* __restrict__ W,
                      const float* __restrict__ bias, void* __restrict__ out,
                      int Mtot, int N, int K, int mode) {
    int lane = threadIdx.x & 31;
    int w    = threadIdx.x >> 5;
    int m0   = blockIdx.x * 32;
    int n0   = (blockIdx.y * 4 + w) * 32;
    if (n0 >= N || m0 >= Mtot) return;

    int n = lane & 15;
    float bv0 = bias[n0 + n];
    float bv1 = bias[n0 + 16 + n];
    v8f c00, c01, c10, c11;
#pragma unroll
    for (int j = 0; j < 8; ++j) { c00[j] = bv0; c01[j] = bv1; c10[j] = bv0; c11[j] = bv1; }

    for (int k0 = 0; k0 < K; k0 += 32) {
        v16h a0 = load_a_f16(A, K, m0,      k0, lane);
        v16h a1 = load_a_f16(A, K, m0 + 16, k0, lane);
        v16h w0 = load_b_f16(W, K, n0,      k0, lane);
        v16h w1 = load_b_f16(W, K, n0 + 16, k0, lane);
        c00 = wmma_f16(a0, w0, c00);
        c01 = wmma_f16(a0, w1, c01);
        c10 = wmma_f16(a1, w0, c10);
        c11 = wmma_f16(a1, w1, c11);
    }

    store_tile(out, c00, m0,      n0,      N, mode, lane);
    store_tile(out, c01, m0,      n0 + 16, N, mode, lane);
    store_tile(out, c10, m0 + 16, n0,      N, mode, lane);
    store_tile(out, c11, m0 + 16, n0 + 16, N, mode, lane);
}

// ---- attention kernel: one block per (b, h, 16-row query tile) ------------
__global__ __launch_bounds__(256)
void attn_kernel(const _Float16* __restrict__ Qh, const _Float16* __restrict__ Kh,
                 const _Float16* __restrict__ Vt, const int* __restrict__ mask,
                 float* __restrict__ scores_out, _Float16* __restrict__ ctx,
                 float scale) {
    extern __shared__ char smem[];
    float* sm  = (float*)smem;                               // [16][LDP] f32 scores
    float* red = (float*)(smem + (size_t)16 * LDP * 4);      // [8][32][8] f32

    int lane = threadIdx.x & 31;
    int w    = threadIdx.x >> 5;
    int qtPer = L_DIM / 16;
    int bid = blockIdx.x;
    int qt = bid % qtPer;
    int h  = (bid / qtPer) % H_DIM;
    int b  = bid / (qtPer * H_DIM);
    int q0 = qt * 16;

    const _Float16* Qb = Qh + (size_t)(b * H_DIM + h) * L_DIM * DK;
    const _Float16* Kb = Kh + (size_t)(b * H_DIM + h) * L_DIM * DK;
    const _Float16* Vb = Vt + (size_t)(b * H_DIM + h) * DK * L_DIM;

    int n = lane & 15, halfm = lane >> 4;

    // --- phase 1: S = Q K^T, masked + scaled, into LDS strip ---
    v16h aq0 = load_a_f16(Qb, DK, q0, 0,  lane);
    v16h aq1 = load_a_f16(Qb, DK, q0, 32, lane);
    for (int nt = w; nt < L_DIM / 16; nt += 8) {
        int k0g = nt * 16;
        if (nt + 8 < L_DIM / 16)   // prefetch next K tile (global_prefetch_b8)
            __builtin_prefetch(Kb + (size_t)(k0g + 128) * DK + (lane << 2), 0, 1);
        v16h b0 = load_b_f16(Kb, DK, k0g, 0,  lane);
        v16h b1 = load_b_f16(Kb, DK, k0g, 32, lane);
        v8f c = {};
        c = wmma_f16(aq0, b0, c);
        c = wmma_f16(aq1, b1, c);
        int kg = k0g + n;
        float am = mask[b * L_DIM + kg] ? 0.0f : NEG_BIG;
#pragma unroll
        for (int j = 0; j < 8; ++j) {
            int M = j + 8 * halfm;
            sm[(size_t)M * LDP + kg] = (c[j] + am) * scale;
        }
    }
    __syncthreads();

    // --- phase 2: row softmax (2 rows per wave), stream probs to global,
    //     and re-pack probs IN PLACE as f16 over the f32 row (f16 row r
    //     overlays the first half of f32 row r; loads precede stores within
    //     the wave, rows are wave-private => no race) ---
    for (int rr = w * 2; rr < w * 2 + 2; ++rr) {
        float*    row  = sm + (size_t)rr * LDP;
        _Float16* hrow = (_Float16*)row;
        float mx = -3.4e38f;
        for (int k = lane; k < L_DIM; k += 32) mx = fmaxf(mx, row[k]);
#pragma unroll
        for (int off = 16; off; off >>= 1) mx = fmaxf(mx, __shfl_xor(mx, off, 32));
        float s = 0.0f;
        for (int k = lane; k < L_DIM; k += 32) {
            float e = __expf(row[k] - mx);
            row[k] = e;
            s += e;
        }
#pragma unroll
        for (int off = 16; off; off >>= 1) s += __shfl_xor(s, off, 32);
        float inv = 1.0f / s;
        float* orow = scores_out + ((size_t)(b * H_DIM + h) * L_DIM + q0 + rr) * L_DIM;
        for (int k = lane; k < L_DIM; k += 32) {
            float p = row[k] * inv;
            hrow[k] = (_Float16)p;   // in-place f16 repack for phase 3
            orow[k] = p;             // streamed attention_scores output
        }
    }
    __syncthreads();

    // --- phase 3: blended = P V, split-K across wave pairs.
    //     P is f16 in LDS with row stride 2*LDP halves -> A-frags are two
    //     aligned 16B ds_load_b128 per WMMA. ---
    const _Float16* Pf16 = (const _Float16*)sm;
    int ntile = w & 3;      // which 16-wide slice of d_k
    int kpart = w >> 2;     // split K=2048 in two halves
    v8f c = {};
    for (int kc = kpart * 32; kc < kpart * 32 + 32; ++kc) {
        int k0 = kc * 32;
        v16h a  = load_a_f16(Pf16, 2 * LDP, 0, k0, lane);
        v16h bv = load_b_f16(Vb, L_DIM, ntile * 16, k0, lane);
        c = wmma_f16(a, bv, c);
    }
#pragma unroll
    for (int j = 0; j < 8; ++j) red[((size_t)w * 32 + lane) * 8 + j] = c[j];
    __syncthreads();
    if (w < 4) {
#pragma unroll
        for (int j = 0; j < 8; ++j) c[j] += red[((size_t)(w + 4) * 32 + lane) * 8 + j];
#pragma unroll
        for (int j = 0; j < 8; ++j) {
            int M = j + 8 * halfm;
            ctx[((size_t)b * L_DIM + q0 + M) * D_MOD + h * DK + ntile * 16 + n] = (_Float16)c[j];
        }
    }
}

// ---------------------------------------------------------------------------
extern "C" void kernel_launch(void* const* d_in, const int* in_sizes, int n_in,
                              void* d_out, int out_size, void* d_ws, size_t ws_size,
                              hipStream_t stream) {
    const float* query = (const float*)d_in[0];
    const float* key   = (const float*)d_in[1];
    const float* value = (const float*)d_in[2];
    const int*   amask = (const int*)  d_in[3];
    const float* wq_w  = (const float*)d_in[4];
    const float* wq_b  = (const float*)d_in[5];
    const float* wk_w  = (const float*)d_in[6];
    const float* wk_b  = (const float*)d_in[7];
    const float* wv_w  = (const float*)d_in[8];
    const float* wv_b  = (const float*)d_in[9];
    const float* wo_w  = (const float*)d_in[10];
    const float* wo_b  = (const float*)d_in[11];

    float* outBlended = (float*)d_out;                               // [B,L,D_EMB]
    float* outScores  = outBlended + (size_t)B_DIM * L_DIM * D_EMB;  // [B,H,L,L]

    const size_t nBLD = (size_t)B_DIM * L_DIM * D_EMB;   // 4,194,304
    const size_t nW   = (size_t)D_MOD * D_EMB;           // 1,048,576

    char* ws = (char*)d_ws;
    _Float16* qx16 = (_Float16*)ws;                ws += nBLD * 2;
    _Float16* kx16 = (_Float16*)ws;                ws += nBLD * 2;
    _Float16* vx16 = (_Float16*)ws;                ws += nBLD * 2;
    _Float16* wq16 = (_Float16*)ws;                ws += nW * 2;
    _Float16* wk16 = (_Float16*)ws;                ws += nW * 2;
    _Float16* wv16 = (_Float16*)ws;                ws += nW * 2;
    _Float16* wo16 = (_Float16*)ws;                ws += nW * 2;
    _Float16* Qh   = (_Float16*)ws;                ws += nBLD * 2;   // [B,H,L,DK]
    _Float16* Kh   = (_Float16*)ws;                ws += nBLD * 2;   // [B,H,L,DK]
    _Float16* Vt   = (_Float16*)ws;                ws += nBLD * 2;   // [B,H,DK,L]
    _Float16* ctx  = (_Float16*)ws;                ws += nBLD * 2;   // [B,L,D_MOD]

    // 1) convert inputs/weights to f16
    {
        int t = 256;
        cvt_f32_f16_kernel<<<(int)((nBLD + t - 1) / t), t, 0, stream>>>(query, qx16, (int)nBLD);
        cvt_f32_f16_kernel<<<(int)((nBLD + t - 1) / t), t, 0, stream>>>(key,   kx16, (int)nBLD);
        cvt_f32_f16_kernel<<<(int)((nBLD + t - 1) / t), t, 0, stream>>>(value, vx16, (int)nBLD);
        cvt_f32_f16_kernel<<<(int)((nW + t - 1) / t),   t, 0, stream>>>(wq_w,  wq16, (int)nW);
        cvt_f32_f16_kernel<<<(int)((nW + t - 1) / t),   t, 0, stream>>>(wk_w,  wk16, (int)nW);
        cvt_f32_f16_kernel<<<(int)((nW + t - 1) / t),   t, 0, stream>>>(wv_w,  wv16, (int)nW);
        cvt_f32_f16_kernel<<<(int)((nW + t - 1) / t),   t, 0, stream>>>(wo_w,  wo16, (int)nW);
    }

    // 2) Q/K/V projections (WMMA, 2x2 blocked)
    {
        dim3 grid(B_DIM * L_DIM / 32, D_MOD / 128);   // (128, 8)
        gemm_proj_kernel<<<grid, 128, 0, stream>>>(qx16, wq16, wq_b, Qh,
                                                   B_DIM * L_DIM, D_MOD, D_EMB, 1);
        gemm_proj_kernel<<<grid, 128, 0, stream>>>(kx16, wk16, wk_b, Kh,
                                                   B_DIM * L_DIM, D_MOD, D_EMB, 1);
        gemm_proj_kernel<<<grid, 128, 0, stream>>>(vx16, wv16, wv_b, Vt,
                                                   B_DIM * L_DIM, D_MOD, D_EMB, 2);
    }

    // 3) attention (scores + softmax + PV)
    {
        int blocks = B_DIM * H_DIM * (L_DIM / 16);   // 4096
        size_t shm = (size_t)16 * LDP * 4 + (size_t)8 * 32 * 8 * 4;  // ~139.5 KB
        float scale = 1.0f / sqrtf((float)L_DIM);
        attn_kernel<<<blocks, 256, shm, stream>>>(Qh, Kh, Vt, amask, outScores, ctx, scale);
    }

    // 4) output projection (WMMA, 2x2 blocked) -> f32 blended
    {
        dim3 grid(B_DIM * L_DIM / 32, D_EMB / 128);
        gemm_proj_kernel<<<grid, 128, 0, stream>>>(ctx, wo16, wo_b, outBlended,
                                                   B_DIM * L_DIM, D_EMB, D_MOD, 0);
    }
}
